// DCAT_57226144252733
// MI455X (gfx1250) — compile-verified, matrix operand order checked by reading, not run on hardware
//
#include <hip/hip_runtime.h>

// ---------------- problem constants ----------------
#define BB  4
#define CC  512
#define NNt 4096     // tokens per batch (H*W)
#define PP  64
#define NHH 8
#define DD  64

typedef __attribute__((ext_vector_type(16))) __bf16 v16bf;
typedef __attribute__((ext_vector_type(8)))  float  v8f;
typedef __attribute__((ext_vector_type(4)))  unsigned int v4u;

__device__ __forceinline__ unsigned short f2bf(float f) {
  unsigned int u = __float_as_uint(f);
  u += 0x7FFFu + ((u >> 16) & 1u);          // RNE
  return (unsigned short)(u >> 16);
}
__device__ __forceinline__ float bf2f(unsigned short h) {
  return __uint_as_float(((unsigned int)h) << 16);
}

// ---------------- generic batched WMMA GEMM (all operands contiguous in K) --
// D(m,j) = sum_k A(m,k)*B(k,j); A/B bf16 (ushort storage), f32 accumulate.
// A(m,k) at A + m*Asm_ + k          (Ask == 1 by construction)
// B(k,j) at B + j*Bsj  + k          (Bsk == 1 by construction)
// Operand base per batch z: base + (z/nh)*bh + (z%nh)*bl.
struct GemmP {
  const unsigned short* A;
  const unsigned short* Bm;
  const float* bias;      // per output column j (or null)
  const float* scale;     // per (z % nh)       (or null)
  const float* res;       // f32 residual       (or null)
  float* Of;              // f32 output         (or null)
  unsigned short* Ob;     // bf16 output        (or null)
  unsigned short* Ob2;    // bf16 shadow output (or null)
  int M, Nn, K, nh, relu;
  long long Asm_, Abh, Abl;
  long long Bsj, Bbh, Bbl;
  long long Osm, Osj, Obh, Obl;
  long long Rsm, Rsj, Rbh, Rbl;
};

union Frag { v16bf v; unsigned short u[16]; v4u q[2]; };

__device__ __forceinline__ void load_frag(Frag& f, const unsigned short* p) {
  const v4u* q = (const v4u*)p;   // 16B-aligned by construction
  f.q[0] = q[0];                  // K offsets 0..7
  f.q[1] = q[2];                  // K offsets 16..23
}

// 8 waves per block: 4 (row tiles) x 2 (col pairs); each wave -> 16x32 output.
__global__ __launch_bounds__(256) void gemm_bf16_wmma(GemmP p) {
  const int lane  = threadIdx.x & 31;
  const int wid   = threadIdx.x >> 5;
  const int row   = lane & 15;
  const int khalf = (lane >> 4) << 3;         // 0 or 8
  const int m0 = blockIdx.y * 64 + (wid >> 1) * 16;
  const int j0 = blockIdx.x * 64 + (wid & 1) * 32;
  if (m0 >= p.M || j0 >= p.Nn) return;
  const int z  = blockIdx.z;
  const int zb = z / p.nh, zh = z % p.nh;

  const unsigned short* Ab = p.A  + (long long)zb * p.Abh + (long long)zh * p.Abl
                                  + (long long)(m0 + row) * p.Asm_ + khalf;
  const unsigned short* Bb = p.Bm + (long long)zb * p.Bbh + (long long)zh * p.Bbl + khalf;
  const unsigned short* B0 = Bb + (long long)(j0 + row) * p.Bsj;
  const unsigned short* B1 = Bb + (long long)(j0 + 16 + row) * p.Bsj;

  v8f acc0 = {0.f, 0.f, 0.f, 0.f, 0.f, 0.f, 0.f, 0.f};
  v8f acc1 = {0.f, 0.f, 0.f, 0.f, 0.f, 0.f, 0.f, 0.f};

  for (int kb = 0; kb < p.K; kb += 32) {
    Frag fa, fb0, fb1;
    load_frag(fa,  Ab + kb);
    load_frag(fb0, B0 + kb);
    load_frag(fb1, B1 + kb);
    acc0 = __builtin_amdgcn_wmma_f32_16x16x32_bf16(
        false, fa.v, false, fb0.v, (short)0, acc0, false, false);
    acc1 = __builtin_amdgcn_wmma_f32_16x16x32_bf16(
        false, fa.v, false, fb1.v, (short)0, acc1, false, false);
  }

  const float s = p.scale ? p.scale[zh] : 1.0f;
  float*          Of  = p.Of  ? p.Of  + (long long)zb * p.Obh + (long long)zh * p.Obl : nullptr;
  unsigned short* Ob  = p.Ob  ? p.Ob  + (long long)zb * p.Obh + (long long)zh * p.Obl : nullptr;
  unsigned short* Ob2 = p.Ob2 ? p.Ob2 + (long long)zb * p.Obh + (long long)zh * p.Obl : nullptr;
  const float*    Rb  = p.res ? p.res + (long long)zb * p.Rbh + (long long)zh * p.Rbl : nullptr;

#pragma unroll
  for (int half = 0; half < 2; ++half) {
    const v8f acc = half ? acc1 : acc0;
    const int j = j0 + half * 16 + row;
#pragma unroll
    for (int r = 0; r < 8; ++r) {
      const int m = m0 + r + khalf;            // C/D layout: lanes 16-31 -> M = r+8
      float v = acc[r] * s;
      if (p.bias) v += p.bias[j];
      if (Rb)     v += Rb[(long long)m * p.Rsm + (long long)j * p.Rsj];
      if (p.relu) v = fmaxf(v, 0.0f);
      const long long oi = (long long)m * p.Osm + (long long)j * p.Osj;
      if (Of)  Of[oi]  = v;
      if (Ob)  Ob[oi]  = f2bf(v);
      if (Ob2) Ob2[oi] = f2bf(v);
    }
  }
}

// ---------------- tokenize + LayerNorm ----------------
// one thread per token: loops over C with fully coalesced [B,C,N] reads
__global__ void tokenize_ln_kernel(const float* __restrict__ x, const float* __restrict__ pos,
                                   const float* __restrict__ g, const float* __restrict__ bta,
                                   float* __restrict__ xt, unsigned short* __restrict__ xn) {
  int t = blockIdx.x * blockDim.x + threadIdx.x;
  if (t >= BB * NNt) return;
  int b = t / NNt, n = t % NNt;
  const float* xb = x + (long long)b * CC * NNt + n;
  float sum = 0.f, ssq = 0.f;
  for (int c = 0; c < CC; ++c) {
    float v = xb[(long long)c * NNt] + pos[(long long)n * CC + c];
    sum += v; ssq += v * v;
  }
  float mu   = sum / (float)CC;
  float var  = ssq / (float)CC - mu * mu;
  float rstd = rsqrtf(var + 1e-5f);
  long long ro = (long long)t * CC;
  for (int c = 0; c < CC; ++c) {
    float v = xb[(long long)c * NNt] + pos[(long long)n * CC + c];
    if (xt) xt[ro + c] = v;
    xn[ro + c] = f2bf((v - mu) * rstd * g[c] + bta[c]);
  }
}

// ------- in-place row L2 normalize (bf16) + optional transposed copy -------
__global__ void l2norm_kernel(unsigned short* d, int len, int rpb,
                              long long bstride, long long rstride,
                              unsigned short* tr, int trpb,
                              long long tbs, long long tstep) {
  __shared__ float red[256];
  int rowid = blockIdx.x;
  unsigned short* p = d + (long long)(rowid / rpb) * bstride
                        + (long long)(rowid % rpb) * rstride;
  float s = 0.f;
  for (int i = threadIdx.x; i < len; i += 256) { float v = bf2f(p[i]); s += v * v; }
  red[threadIdx.x] = s; __syncthreads();
  for (int st = 128; st > 0; st >>= 1) {
    if (threadIdx.x < st) red[threadIdx.x] += red[threadIdx.x + st];
    __syncthreads();
  }
  float scale = 1.0f / fmaxf(sqrtf(red[0]), 1e-12f);
  if (tr) {
    unsigned short* tp = tr + (long long)(rowid / trpb) * tbs + (long long)(rowid % trpb);
    for (int i = threadIdx.x; i < len; i += 256) {
      unsigned short v = f2bf(bf2f(p[i]) * scale);
      p[i] = v;
      tp[(long long)i * tstep] = v;
    }
  } else {
    for (int i = threadIdx.x; i < len; i += 256) p[i] = f2bf(bf2f(p[i]) * scale);
  }
}

// ---------------- softmax over rows of 64 (f32 in, bf16 out) ----------------
__global__ void softmax64_kernel(const float* __restrict__ in, unsigned short* __restrict__ out) {
  __shared__ float buf[64];
  long long r = blockIdx.x;
  int t = threadIdx.x;
  float v = in[r * 64 + t];
  buf[t] = v; __syncthreads();
  for (int s = 32; s > 0; s >>= 1) { if (t < s) buf[t] = fmaxf(buf[t], buf[t + s]); __syncthreads(); }
  float mx = buf[0]; __syncthreads();
  float e = __expf(v - mx);
  buf[t] = e; __syncthreads();
  for (int s = 32; s > 0; s >>= 1) { if (t < s) buf[t] += buf[t + s]; __syncthreads(); }
  out[r * 64 + t] = f2bf(e / buf[0]);
}

// ------- f32 [K,Nc] row-major -> bf16 [Nc,K] (transposed) -------
__global__ void f2bf_t_kernel(const float* __restrict__ src, unsigned short* __restrict__ dst,
                              int K, int Nc) {
  long long t = (long long)blockIdx.x * blockDim.x + threadIdx.x;
  if (t >= (long long)K * Nc) return;
  int j = (int)(t / K), k = (int)(t % K);
  dst[t] = f2bf(src[(long long)k * Nc + j]);
}

// ---------------- host launcher ----------------
extern "C" void kernel_launch(void* const* d_in, const int* in_sizes, int n_in,
                              void* d_out, int out_size, void* d_ws, size_t ws_size,
                              hipStream_t stream) {
  (void)in_sizes; (void)n_in; (void)out_size; (void)ws_size;
  const float* x    = (const float*)d_in[0];
  const float* ref  = (const float*)d_in[1];
  const float* pos  = (const float*)d_in[2];
  const float* ln_g = (const float*)d_in[3];
  const float* ln_b = (const float*)d_in[4];
  const float* Wq   = (const float*)d_in[5];
  const float* bq   = (const float*)d_in[6];
  const float* Wkvv = (const float*)d_in[7];
  const float* bkvv = (const float*)d_in[8];
  const float* We   = (const float*)d_in[9];
  const float* be   = (const float*)d_in[10];
  const float* temp  = (const float*)d_in[11];
  const float* temp2 = (const float*)d_in[12];
  const float* Wo1  = (const float*)d_in[13];
  const float* bo1  = (const float*)d_in[14];
  const float* Wo2  = (const float*)d_in[15];
  const float* bo2  = (const float*)d_in[16];
  const float* Wf1  = (const float*)d_in[17];
  const float* bf1  = (const float*)d_in[18];
  const float* Wf2  = (const float*)d_in[19];
  const float* bf2v = (const float*)d_in[20];
  float* out = (float*)d_out;

  char* ws = (char*)d_ws;
  size_t off = 0;
  auto alloc = [&](size_t bytes) -> void* {
    off = (off + 255) & ~(size_t)255;
    void* p = ws + off;
    off += bytes;
    return p;
  };
  const long long BNC = (long long)BB * NNt * CC;

  float*          xt_f   = (float*)         alloc(BNC * 4);     // tokens + pos (residual)
  unsigned short* xn_b   = (unsigned short*)alloc(BNC * 2);     // LN(x)   bf16 [B,N,C]
  unsigned short* rn_b   = (unsigned short*)alloc(BNC * 2);     // LN(ref) bf16 [B,N,C]
  unsigned short* qb     = (unsigned short*)alloc(BNC * 2);     // q->qn   [B,C,N]
  unsigned short* qt     = (unsigned short*)alloc(BNC * 2);     // qn      [B,N,C]
  unsigned short* kb     = (unsigned short*)alloc(BNC * 2);     // k->kn   [B,C,N]
  unsigned short* vca    = (unsigned short*)alloc(BNC * 2);     // v_ca    [B,N,C]
  unsigned short* vsa    = (unsigned short*)alloc(BNC * 2);     // v_sa    [B,C,N]
  float*          S32    = (float*)         alloc((long long)32 * DD * DD * 4);
  unsigned short* Sbf    = (unsigned short*)alloc((long long)32 * DD * DD * 2);
  unsigned short* xca    = (unsigned short*)alloc(BNC * 2);     // [B,N,C]
  unsigned short* kprojt = (unsigned short*)alloc((long long)32 * DD * PP * 2); // [32,P,D]
  unsigned short* vproj  = (unsigned short*)alloc((long long)32 * DD * PP * 2); // [32,D,P]
  float*          SA32   = (float*)         alloc((long long)32 * NNt * PP * 4);
  unsigned short* SAbf   = (unsigned short*)alloc((long long)32 * NNt * PP * 2);
  unsigned short* xsa    = (unsigned short*)alloc(BNC * 2);     // bug-faithful [B,N,C] view
  float*          attn_f = (float*)         alloc(BNC * 4);
  unsigned short* attnbf = (unsigned short*)alloc(BNC * 2);
  unsigned short* h1     = (unsigned short*)alloc(BNC * 2);
  // transposed bf16 weights: Wt[j,k] = W[k,j]
  unsigned short* Wq_t   = (unsigned short*)alloc((long long)CC * CC * 2);
  unsigned short* Wkvv_t = (unsigned short*)alloc((long long)CC * 3 * CC * 2);
  unsigned short* We_t   = (unsigned short*)alloc((long long)NNt * PP * 2);
  unsigned short* Wo1_t  = (unsigned short*)alloc((long long)CC * (CC / 2) * 2);
  unsigned short* Wo2_t  = (unsigned short*)alloc((long long)CC * (CC / 2) * 2);
  unsigned short* Wf1_t  = (unsigned short*)alloc((long long)CC * CC * 2);
  unsigned short* Wf2_t  = (unsigned short*)alloc((long long)CC * CC * 2);

  auto convT = [&](const float* src, unsigned short* dst, int K, int Nc) {
    long long n = (long long)K * Nc;
    f2bf_t_kernel<<<dim3((unsigned)((n + 255) / 256)), dim3(256), 0, stream>>>(src, dst, K, Nc);
  };
  convT(Wq,   Wq_t,   CC, CC);
  convT(Wkvv, Wkvv_t, CC, 3 * CC);
  convT(We,   We_t,   NNt, PP);
  convT(Wo1,  Wo1_t,  CC, CC / 2);
  convT(Wo2,  Wo2_t,  CC, CC / 2);
  convT(Wf1,  Wf1_t,  CC, CC);
  convT(Wf2,  Wf2_t,  CC, CC);

  // tokenize + LN
  tokenize_ln_kernel<<<dim3((BB * NNt) / 256), dim3(256), 0, stream>>>(x,   pos, ln_g, ln_b, xt_f, xn_b);
  tokenize_ln_kernel<<<dim3((BB * NNt) / 256), dim3(256), 0, stream>>>(ref, pos, ln_g, ln_b, nullptr, rn_b);

  const dim3 blk(256);
  auto grid = [](int Nn, int M, int z) { return dim3((unsigned)(Nn / 64), (unsigned)(M / 64), (unsigned)z); };

  // q = xn @ Wq + bq -> qb [B,C,N]
  { GemmP p{}; p.A = xn_b; p.Bm = Wq_t; p.bias = bq; p.Ob = qb;
    p.M = NNt; p.Nn = CC; p.K = CC; p.nh = 1;
    p.Asm_ = CC; p.Abh = (long long)NNt * CC;
    p.Bsj = CC;
    p.Osm = 1; p.Osj = NNt; p.Obh = (long long)CC * NNt;
    gemm_bf16_wmma<<<grid(CC, NNt, BB), blk, 0, stream>>>(p); }
  // kvv split: k -> kb [B,C,N]; v_ca -> vca [B,N,C]; v_sa -> vsa [B,C,N]
  { GemmP p{}; p.A = rn_b; p.M = NNt; p.Nn = CC; p.K = CC; p.nh = 1;
    p.Asm_ = CC; p.Abh = (long long)NNt * CC;
    p.Bsj = CC;
    // k
    p.Bm = Wkvv_t;                 p.bias = bkvv;          p.Ob = kb;
    p.Osm = 1; p.Osj = NNt; p.Obh = (long long)CC * NNt;
    gemm_bf16_wmma<<<grid(CC, NNt, BB), blk, 0, stream>>>(p);
    // v_ca (token-major)
    p.Bm = Wkvv_t + 1LL * CC * CC; p.bias = bkvv + CC;     p.Ob = vca;
    p.Osm = CC; p.Osj = 1; p.Obh = (long long)NNt * CC;
    gemm_bf16_wmma<<<grid(CC, NNt, BB), blk, 0, stream>>>(p);
    // v_sa
    p.Bm = Wkvv_t + 2LL * CC * CC; p.bias = bkvv + 2 * CC; p.Ob = vsa;
    p.Osm = 1; p.Osj = NNt; p.Obh = (long long)CC * NNt;
    gemm_bf16_wmma<<<grid(CC, NNt, BB), blk, 0, stream>>>(p); }

  // l2norm over N: qn in place + transposed copy qt [B,N,C]; kn in place
  l2norm_kernel<<<dim3(BB * CC), dim3(256), 0, stream>>>(
      qb, NNt, BB * CC, 0, (long long)NNt, qt, CC, (long long)NNt * CC, (long long)CC);
  l2norm_kernel<<<dim3(BB * CC), dim3(256), 0, stream>>>(
      kb, NNt, BB * CC, 0, (long long)NNt, nullptr, 1, 0, 0);

  // channel scores: S = (qn_bh @ kn_bh^T) * temp[h]   (z = b*8+h)
  { GemmP p{}; p.A = qb; p.Bm = kb; p.scale = temp; p.Of = S32;
    p.M = DD; p.Nn = DD; p.K = NNt; p.nh = NHH;
    p.Asm_ = NNt; p.Abh = (long long)CC * NNt; p.Abl = (long long)DD * NNt;
    p.Bsj = NNt;  p.Bbh = (long long)CC * NNt; p.Bbl = (long long)DD * NNt;
    p.Osm = DD; p.Osj = 1; p.Obh = 8LL * DD * DD; p.Obl = (long long)DD * DD;
    gemm_bf16_wmma<<<grid(DD, DD, 32), blk, 0, stream>>>(p); }
  softmax64_kernel<<<dim3(32 * DD), dim3(64), 0, stream>>>(S32, Sbf);

  // x_ca = S @ v_ca -> [B,N,C] (out[b,n,h*D+d]); B from token-major vca: Bsk=1
  { GemmP p{}; p.A = Sbf; p.Bm = vca; p.Ob = xca;
    p.M = DD; p.Nn = NNt; p.K = DD; p.nh = NHH;
    p.Asm_ = DD; p.Abh = 8LL * DD * DD; p.Abl = (long long)DD * DD;
    p.Bsj = CC;  p.Bbh = (long long)NNt * CC; p.Bbl = DD;
    p.Osm = 1; p.Osj = CC; p.Obh = (long long)NNt * CC; p.Obl = DD;
    gemm_bf16_wmma<<<grid(NNt, DD, 32), blk, 0, stream>>>(p); }

  // k_proj^T [32,P,D] and v_proj [32,D,P] via We_t [P,N]
  { GemmP p{}; p.A = kb; p.Bm = We_t; p.bias = be; p.Ob = kprojt;
    p.M = DD; p.Nn = PP; p.K = NNt; p.nh = NHH;
    p.Asm_ = NNt; p.Abh = (long long)CC * NNt; p.Abl = (long long)DD * NNt;
    p.Bsj = NNt;
    p.Osm = 1; p.Osj = DD; p.Obh = 8LL * DD * PP; p.Obl = (long long)DD * PP;
    gemm_bf16_wmma<<<grid(PP, DD, 32), blk, 0, stream>>>(p);
    p.A = vsa; p.Ob = vproj;
    p.Osm = PP; p.Osj = 1;   // row-major [D,P]
    gemm_bf16_wmma<<<grid(PP, DD, 32), blk, 0, stream>>>(p); }

  // Linformer scores: (qn^T @ k_proj) * temp2[h] -> SA32 [32,N,P]
  { GemmP p{}; p.A = qt; p.Bm = kprojt; p.scale = temp2; p.Of = SA32;
    p.M = NNt; p.Nn = PP; p.K = DD; p.nh = NHH;
    p.Asm_ = CC; p.Abh = (long long)NNt * CC; p.Abl = DD;
    p.Bsj = DD;  p.Bbh = 8LL * DD * PP; p.Bbl = (long long)DD * PP;
    p.Osm = PP; p.Osj = 1; p.Obh = 8LL * NNt * PP; p.Obl = (long long)NNt * PP;
    gemm_bf16_wmma<<<grid(PP, NNt, 32), blk, 0, stream>>>(p); }
  softmax64_kernel<<<dim3(32 * NNt), dim3(64), 0, stream>>>(SA32, SAbf);

  // x_sa = attn_sa @ v_proj^T, bug-faithful scatter: flat f = d*NH*N + h*N + n
  { GemmP p{}; p.A = SAbf; p.Bm = vproj; p.Ob = xsa;
    p.M = NNt; p.Nn = DD; p.K = PP; p.nh = NHH;
    p.Asm_ = PP; p.Abh = 8LL * NNt * PP; p.Abl = (long long)NNt * PP;
    p.Bsj = PP;  p.Bbh = 8LL * DD * PP;  p.Bbl = (long long)DD * PP;
    p.Osm = 1; p.Osj = (long long)NHH * NNt; p.Obh = (long long)NNt * CC; p.Obl = NNt;
    gemm_bf16_wmma<<<grid(DD, NNt, 32), blk, 0, stream>>>(p); }

  // out-proj + concat + residual(xt): attn = xt + [xsa@Wo2+bo2 | xca@Wo1+bo1]
  { GemmP p{}; p.A = xsa; p.Bm = Wo2_t; p.bias = bo2; p.res = xt_f;
    p.Of = attn_f; p.Ob2 = attnbf;
    p.M = NNt; p.Nn = CC / 2; p.K = CC; p.nh = 1;
    p.Asm_ = CC; p.Abh = (long long)NNt * CC;
    p.Bsj = CC;
    p.Osm = CC; p.Osj = 1; p.Obh = (long long)NNt * CC;
    p.Rsm = CC; p.Rsj = 1; p.Rbh = (long long)NNt * CC;
    gemm_bf16_wmma<<<grid(CC / 2, NNt, BB), blk, 0, stream>>>(p);
    p.A = xca; p.Bm = Wo1_t; p.bias = bo1;
    p.res = xt_f + CC / 2; p.Of = attn_f + CC / 2; p.Ob2 = attnbf + CC / 2;
    gemm_bf16_wmma<<<grid(CC / 2, NNt, BB), blk, 0, stream>>>(p); }

  // FFN1: h1 = relu(attn @ Wf1 + bf1)
  { GemmP p{}; p.A = attnbf; p.Bm = Wf1_t; p.bias = bf1; p.Ob = h1; p.relu = 1;
    p.M = BB * NNt; p.Nn = CC; p.K = CC; p.nh = 1;
    p.Asm_ = CC;
    p.Bsj = CC;
    p.Osm = CC; p.Osj = 1;
    gemm_bf16_wmma<<<grid(CC, BB * NNt, 1), blk, 0, stream>>>(p); }

  // FFN2 + residual, write final output in [B,C,H,W] layout
  { GemmP p{}; p.A = h1; p.Bm = Wf2_t; p.bias = bf2v; p.res = attn_f; p.Of = out;
    p.M = NNt; p.Nn = CC; p.K = CC; p.nh = 1;
    p.Asm_ = CC; p.Abh = (long long)NNt * CC;
    p.Bsj = CC;
    p.Osm = 1; p.Osj = NNt; p.Obh = (long long)CC * NNt;
    p.Rsm = CC; p.Rsj = 1; p.Rbh = (long long)NNt * CC;
    gemm_bf16_wmma<<<grid(CC, NNt, BB), blk, 0, stream>>>(p); }
}
